// EfficientDeformConv1d_8546984919348
// MI455X (gfx1250) — matrix operand order repaired and checked by reading, not compiled, not run on hardware
//
#include <hip/hip_runtime.h>
#include <stdint.h>

// ---------------------------------------------------------------------------
// Deformable Conv1d for MI455X (gfx1250): both heavy stages are GEMMs done
// with v_wmma_f32_16x16x32_bf16 (fp32 accumulate). Sampling/interp fused into
// the B-operand staging of the main GEMM. This revision packs bf16 pairs for
// dword LDS stores and uses 32-bit gather offsets (saddr+voffset addressing).
// ---------------------------------------------------------------------------

typedef __attribute__((ext_vector_type(16))) __bf16 v16bf;
typedef __attribute__((ext_vector_type(8)))  float  v8f;

#define B_SZ  8
#define C_IN  256
#define C_OUT 256
#define L_LEN 4096
#define KW    3
#define KDIM  (KW * C_IN)   // 768, reordered as j = k*256 + c

__device__ __forceinline__ unsigned int f2bf_bits(float f) {
    unsigned int u = __float_as_uint(f);
    u += 0x7FFFu + ((u >> 16) & 1u);
    return u >> 16;
}
__device__ __forceinline__ unsigned int f2bf2(float lo, float hi) {
    return f2bf_bits(lo) | (f2bf_bits(hi) << 16);
}

union Frag {
    v16bf v;
    uint4 q[2];
};

// ---------------------------------------------------------------------------
// Kernel 0: repack fp32 weights [co][ci][k] -> bf16 [co][k*256+ci]
// ---------------------------------------------------------------------------
__global__ void pack_weights_kernel(const float* __restrict__ offset_w,
                                    const float* __restrict__ weight,
                                    unsigned short* __restrict__ wob,
                                    unsigned short* __restrict__ wb) {
    int idx = blockIdx.x * blockDim.x + threadIdx.x;
    const int total = C_OUT * C_IN * KW;
    if (idx >= total) return;
    int k  = idx % KW;
    int ci = (idx / KW) % C_IN;
    int co = idx / (KW * C_IN);
    int dst = co * KDIM + k * C_IN + ci;
    wob[dst] = (unsigned short)f2bf_bits(offset_w[idx]);
    wb[dst]  = (unsigned short)f2bf_bits(weight[idx]);
}

// ---------------------------------------------------------------------------
// Kernel 1: offset-field conv as bf16 WMMA GEMM.
//   off[b,co,l] = sum_{ci,k} W[co, k*256+ci] * x[b, ci, l-1+k]  + offset_b[co]
// WG tile: 128 (co) x 128 (l); 8 waves, each 64x32 (4x2 WMMA frags).
// ---------------------------------------------------------------------------
__launch_bounds__(256)
__global__ void offset_conv_kernel(const float* __restrict__ x,
                                   const unsigned short* __restrict__ wob,
                                   const float* __restrict__ offset_b,
                                   float* __restrict__ off) {
    __shared__ __align__(16) unsigned short As[128 * 32]; // [row][K]  8 KB
    __shared__ __align__(16) unsigned short Bs[128 * 32]; // [col][K]  8 KB

    const int l0 = blockIdx.x * 128;
    const int o0 = blockIdx.y * 128;
    const int b  = blockIdx.z;

    const int tid  = threadIdx.x;
    const int lane = tid & 31;
    const int wid  = tid >> 5;
    const int wm   = wid & 1;    // 64-row group
    const int wn   = wid >> 1;   // 32-col group
    const int half = lane >> 4;  // K half-split per CDNA5 16-bit layout
    const int ln   = lane & 15;

    v8f zero = {};
    v8f acc[4][2];
    #pragma unroll
    for (int i = 0; i < 4; ++i)
        #pragma unroll
        for (int j = 0; j < 2; ++j) acc[i][j] = zero;

    const float* xb = x + (size_t)b * C_IN * L_LEN;   // 4 MB slice: 32-bit idx ok

    for (int j0 = 0; j0 < KDIM; j0 += 32) {
        const int kk = j0 >> 8;    // kernel tap 0..2
        const int c0 = j0 & 255;   // channel base

        // --- A tile: wob rows [o0,o0+128), K cols [j0,j0+32) (dword loads) ---
        {
            const unsigned int* wg = (const unsigned int*)wob;
            unsigned int* as = (unsigned int*)As;
            #pragma unroll
            for (int i = 0; i < 8; ++i) {
                unsigned int u = tid + i * 256;      // 0..2047 dwords
                unsigned int row = u >> 4;
                unsigned int cp  = u & 15;
                as[u] = wg[(o0 + row) * (KDIM / 2) + (unsigned)(j0 >> 1) + cp];
            }
            if (j0 + 32 < KDIM) {   // prefetch next K-step's weight tile
                unsigned int row = tid >> 1;
                __builtin_prefetch(&wg[(o0 + row) * (KDIM / 2) +
                                       (unsigned)((j0 + 32) >> 1)], 0, 0);
            }
        }
        // --- B tile: Bs[l][cc] = bf16( x[b, c0+cc, l0+l-1+kk] ), zero-pad;
        //     two channels packed per dword store ---
        {
            const int l  = tid & 127;
            const int ps = tid >> 7;           // pair-start 0/1
            const int gx = l0 + l - 1 + kk;
            const bool inb = (gx >= 0) && (gx < L_LEN);
            unsigned int* bsw = (unsigned int*)Bs;
            #pragma unroll
            for (int i = 0; i < 8; ++i) {
                int p  = ps + 2 * i;           // pair index 0..15
                int cc = 2 * p;
                float v0 = 0.0f, v1 = 0.0f;
                if (inb) {
                    unsigned int base = (unsigned)(c0 + cc) * L_LEN + (unsigned)gx;
                    v0 = xb[base];
                    v1 = xb[base + L_LEN];
                }
                bsw[l * 16 + p] = f2bf2(v0, v1);
            }
        }
        __syncthreads();

        Frag bfr[2];
        #pragma unroll
        for (int nt = 0; nt < 2; ++nt) {
            int col = wn * 32 + nt * 16 + ln;
            bfr[nt].q[0] = *(const uint4*)&Bs[col * 32 + half * 8];
            bfr[nt].q[1] = *(const uint4*)&Bs[col * 32 + 16 + half * 8];
        }
        #pragma unroll
        for (int mt = 0; mt < 4; ++mt) {
            Frag afr;
            int row = wm * 64 + mt * 16 + ln;
            afr.q[0] = *(const uint4*)&As[row * 32 + half * 8];
            afr.q[1] = *(const uint4*)&As[row * 32 + 16 + half * 8];
            #pragma unroll
            for (int nt = 0; nt < 2; ++nt) {
                acc[mt][nt] = __builtin_amdgcn_wmma_f32_16x16x32_bf16(
                    false, afr.v, false, bfr[nt].v, (short)0, acc[mt][nt],
                    false, false);
            }
        }
        __syncthreads();
    }

    // --- store (C/D layout: M = i + 8*half, N = lane&15) + bias ---
    #pragma unroll
    for (int mt = 0; mt < 4; ++mt) {
        int ob = o0 + wm * 64 + mt * 16;
        #pragma unroll
        for (int nt = 0; nt < 2; ++nt) {
            int l = l0 + wn * 32 + nt * 16 + ln;
            #pragma unroll
            for (int i = 0; i < 8; ++i) {
                int row = ob + i + half * 8;
                off[((size_t)b * C_IN + row) * L_LEN + l] =
                    acc[mt][nt][i] + offset_b[row];
            }
        }
    }
}

// ---------------------------------------------------------------------------
// Kernel 2: fused deformable sampling + main GEMM (bf16 WMMA).
//   sampled[b,c,k,l] = lerp(x[b,c,:], l-1+k+off[b,c,l])   (zero OOB)
//   out[b,o,l] = sum_{c,k} W[o, k*256+c] * sampled + bias[o]
// ---------------------------------------------------------------------------
__launch_bounds__(256)
__global__ void deform_gemm_kernel(const float* __restrict__ x,
                                   const unsigned short* __restrict__ wb,
                                   const float* __restrict__ bias,
                                   const float* __restrict__ off,
                                   float* __restrict__ out) {
    __shared__ __align__(16) unsigned short As[128 * 32];
    __shared__ __align__(16) unsigned short Bs[128 * 32];

    const int l0 = blockIdx.x * 128;
    const int o0 = blockIdx.y * 128;
    const int b  = blockIdx.z;

    const int tid  = threadIdx.x;
    const int lane = tid & 31;
    const int wid  = tid >> 5;
    const int wm   = wid & 1;
    const int wn   = wid >> 1;
    const int half = lane >> 4;
    const int ln   = lane & 15;

    v8f zero = {};
    v8f acc[4][2];
    #pragma unroll
    for (int i = 0; i < 4; ++i)
        #pragma unroll
        for (int j = 0; j < 2; ++j) acc[i][j] = zero;

    const float* xb   = x   + (size_t)b * C_IN * L_LEN;  // 4 MB slice
    const float* offb = off + (size_t)b * C_IN * L_LEN;  // 4 MB slice

    for (int j0 = 0; j0 < KDIM; j0 += 32) {
        const int kk = j0 >> 8;
        const int c0 = j0 & 255;

        // --- A tile from repacked main weights ---
        {
            const unsigned int* wg = (const unsigned int*)wb;
            unsigned int* as = (unsigned int*)As;
            #pragma unroll
            for (int i = 0; i < 8; ++i) {
                unsigned int u = tid + i * 256;
                unsigned int row = u >> 4;
                unsigned int cp  = u & 15;
                as[u] = wg[(o0 + row) * (KDIM / 2) + (unsigned)(j0 >> 1) + cp];
            }
            if (j0 + 32 < KDIM) {
                unsigned int row = tid >> 1;
                __builtin_prefetch(&wg[(o0 + row) * (KDIM / 2) +
                                       (unsigned)((j0 + 32) >> 1)], 0, 0);
            }
        }
        // --- B tile: deformable sample (fp32 gather + lerp) -> packed bf16 ---
        {
            const int l  = tid & 127;
            const int ps = tid >> 7;
            const int gl = l0 + l;
            unsigned int* bsw = (unsigned int*)Bs;
            #pragma unroll
            for (int i = 0; i < 8; ++i) {
                int p  = ps + 2 * i;
                int cc = 2 * p;
                float s[2];
                #pragma unroll
                for (int q = 0; q < 2; ++q) {
                    unsigned int c = (unsigned)(c0 + cc + q);
                    float ofv = offb[c * L_LEN + (unsigned)gl];
                    float pos = (float)(gl - 1 + kk) + ofv;
                    float p0  = floorf(pos);
                    float w1  = pos - p0;
                    int i0 = (int)p0;
                    int i1 = i0 + 1;
                    unsigned int rb = c * L_LEN;
                    float v0 = (i0 >= 0 && i0 < L_LEN) ? xb[rb + (unsigned)i0] : 0.0f;
                    float v1 = (i1 >= 0 && i1 < L_LEN) ? xb[rb + (unsigned)i1] : 0.0f;
                    s[q] = v0 * (1.0f - w1) + v1 * w1;
                }
                bsw[l * 16 + p] = f2bf2(s[0], s[1]);
            }
        }
        __syncthreads();

        Frag bfr[2];
        #pragma unroll
        for (int nt = 0; nt < 2; ++nt) {
            int col = wn * 32 + nt * 16 + ln;
            bfr[nt].q[0] = *(const uint4*)&Bs[col * 32 + half * 8];
            bfr[nt].q[1] = *(const uint4*)&Bs[col * 32 + 16 + half * 8];
        }
        #pragma unroll
        for (int mt = 0; mt < 4; ++mt) {
            Frag afr;
            int row = wm * 64 + mt * 16 + ln;
            afr.q[0] = *(const uint4*)&As[row * 32 + half * 8];
            afr.q[1] = *(const uint4*)&As[row * 32 + 16 + half * 8];
            #pragma unroll
            for (int nt = 0; nt < 2; ++nt) {
                acc[mt][nt] = __builtin_amdgcn_wmma_f32_16x16x32_bf16(
                    false, afr.v, false, bfr[nt].v, (short)0, acc[mt][nt],
                    false, false);
            }
        }
        __syncthreads();
    }

    #pragma unroll
    for (int mt = 0; mt < 4; ++mt) {
        int ob = o0 + wm * 64 + mt * 16;
        #pragma unroll
        for (int nt = 0; nt < 2; ++nt) {
            int l = l0 + wn * 32 + nt * 16 + ln;
            #pragma unroll
            for (int i = 0; i < 8; ++i) {
                int row = ob + i + half * 8;
                out[((size_t)b * C_OUT + row) * L_LEN + l] =
                    acc[mt][nt][i] + bias[row];
            }
        }
    }
}

// ---------------------------------------------------------------------------
extern "C" void kernel_launch(void* const* d_in, const int* in_sizes, int n_in,
                              void* d_out, int out_size, void* d_ws, size_t ws_size,
                              hipStream_t stream) {
    const float* x        = (const float*)d_in[0];
    const float* offset_w = (const float*)d_in[1];
    const float* offset_b = (const float*)d_in[2];
    const float* weight   = (const float*)d_in[3];
    const float* bias     = (const float*)d_in[4];

    // workspace layout: off (fp32, 33.5 MB) | wob (bf16) | wb (bf16)
    char* ws = (char*)d_ws;
    float* off = (float*)ws;
    size_t off_bytes = (size_t)B_SZ * C_IN * L_LEN * sizeof(float);
    unsigned short* wob = (unsigned short*)(ws + off_bytes);
    unsigned short* wbp = wob + (size_t)C_OUT * KDIM;

    const int n_w = C_OUT * C_IN * KW;
    pack_weights_kernel<<<(n_w + 255) / 256, 256, 0, stream>>>(
        offset_w, weight, wob, wbp);

    dim3 grid(L_LEN / 128, C_OUT / 128, B_SZ);
    offset_conv_kernel<<<grid, 256, 0, stream>>>(x, wob, offset_b, off);
    deform_gemm_kernel<<<grid, 256, 0, stream>>>(x, wbp, bias, off,
                                                 (float*)d_out);
}
